// SparseMoE_15822659518959
// MI455X (gfx1250) — matrix operand order repaired and checked by reading, not compiled, not run on hardware
//
#include <hip/hip_runtime.h>
#include <hip/hip_bf16.h>
#include <cstdint>
#include <cstddef>

// ---------------- problem constants ----------------
#define T_TOK 4096      // B*S tokens
#define DDIM  1024
#define EEXP  8
#define KTOP  2
#define FDIM  4096

// ---------------- tiling ----------------
#define BM 128
#define BN 64
#define BK 32
#define TILES_M (T_TOK / BM)   // 32

typedef __attribute__((ext_vector_type(16))) __bf16 v16bf;
typedef __attribute__((ext_vector_type(8)))  __bf16 v8bf;
typedef __attribute__((ext_vector_type(8)))  float  v8f;
typedef __attribute__((ext_vector_type(4)))  unsigned int v4u;
typedef __attribute__((ext_vector_type(8)))  int v8i;
typedef __attribute__((ext_vector_type(4)))  int v4i;

// ---------------- workspace layout (bytes) ----------------
// xbf  : [T,D]  bf16                         =  8 MiB
// hbuf : [T*K,F] bf16 (compact gathered)     = 64 MiB
// w1t  : [E,F,D] bf16 (n-major transposed)   = 64 MiB
// w3t  : [E,F,D] bf16                        = 64 MiB
// w2t  : [E,D,F] bf16 (n-major transposed)   = 64 MiB
// counters / offsets / routing lists
static constexpr size_t WS_XBF   = 0;
static constexpr size_t WS_HBUF  = WS_XBF  + (size_t)T_TOK * DDIM * 2;
static constexpr size_t WS_W1T   = WS_HBUF + (size_t)T_TOK * KTOP * FDIM * 2;
static constexpr size_t WS_W3T   = WS_W1T  + (size_t)EEXP * DDIM * FDIM * 2;
static constexpr size_t WS_W2T   = WS_W3T  + (size_t)EEXP * DDIM * FDIM * 2;
static constexpr size_t WS_CNT   = WS_W2T  + (size_t)EEXP * DDIM * FDIM * 2;
static constexpr size_t WS_OFF   = WS_CNT  + 64;
static constexpr size_t WS_LIST  = WS_OFF  + 64;
static constexpr size_t WS_SLOTW = WS_LIST + (size_t)EEXP * T_TOK * 4;

// ---------------- TDM helper: 2D bf16 tile Global -> LDS ----------------
// D# per CDNA5 ISA. Group0: count=1 | lds_addr |
// global_addr[56:0] | type=2. Group1: data_size=2B, tensor_dim0/1 (elements),
// tile_dim0 (contiguous X), tile_dim1 (rows), tensor_dim0_stride (elements).
// LDS receives tile rows packed contiguously: exactly our [rows][BK] layout.
// This toolchain exposes the 6-arg builtin: (g0, g1, g2, g3, g_extra, cpol).
static __device__ inline void tdm_load_2d(unsigned lds_addr, const void* gptr,
                                          unsigned tensor_d0, unsigned tensor_d1,
                                          unsigned tile_d0, unsigned tile_d1,
                                          unsigned stride_d0) {
  const unsigned long long ga = (unsigned long long)(uintptr_t)gptr;
  v4u g0;
  g0[0] = 1u;                                            // count=1 (valid user D#)
  g0[1] = lds_addr;                                      // LDS byte address
  g0[2] = (unsigned)ga;                                  // global_addr[31:0]
  g0[3] = (unsigned)((ga >> 32) & 0x01FFFFFFu) | 0x80000000u; // [56:32] | type=2
  v8i g1;
  g1[0] = (int)(1u << 16);                               // wg_mask=0, data_size=2B
  g1[1] = (int)((tensor_d0 & 0xFFFFu) << 16);            // tensor_dim0[15:0]
  g1[2] = (int)((tensor_d0 >> 16) | ((tensor_d1 & 0xFFFFu) << 16));
  g1[3] = (int)((tensor_d1 >> 16) | (tile_d0 << 16));    // tile_dim0
  g1[4] = (int)(tile_d1 & 0xFFFFu);                      // tile_dim1, tile_dim2=0
  g1[5] = (int)stride_d0;                                // dim0_stride[31:0]
  g1[6] = 0;
  g1[7] = 0;
  const v4i z4 = {0, 0, 0, 0};
  const v8i z8 = {0, 0, 0, 0, 0, 0, 0, 0};
  __builtin_amdgcn_tensor_load_to_lds(g0, g1, z4, z4, z8, 0);
}

static __device__ inline unsigned lds_byte_addr(const void* p) {
  // Flat LDS aperture: addr[31:0] is the wave-relative LDS byte address.
  return (unsigned)(uintptr_t)p;
}

// ---------------- kernel 0: zero output + counters ----------------
__global__ void k_zero(float* __restrict__ out, int n, int* __restrict__ counts) {
  for (int i = blockIdx.x * blockDim.x + threadIdx.x; i < n;
       i += gridDim.x * blockDim.x)
    out[i] = 0.0f;
  if (blockIdx.x == 0 && threadIdx.x < EEXP) counts[threadIdx.x] = 0;
}

// ---------------- kernel P: fp32 [R,C] -> bf16 [C,R] (cast + transpose) -----
__global__ __launch_bounds__(256)
void k_transpose_cast(const float* __restrict__ src, __bf16* __restrict__ dst,
                      int R, int C) {
  __shared__ float t[32][33];
  const int bx = blockIdx.x * 32;            // col tile
  const int by = blockIdx.y * 32;            // row tile
  const size_t eo = (size_t)blockIdx.z * R * C;
  src += eo; dst += eo;
  const int tx = threadIdx.x, ty = threadIdx.y;     // (32, 8)
#pragma unroll
  for (int i = ty; i < 32; i += 8)
    t[i][tx] = src[(size_t)(by + i) * C + bx + tx];
  __syncthreads();
#pragma unroll
  for (int i = ty; i < 32; i += 8)
    dst[(size_t)(bx + i) * R + by + tx] = (__bf16)t[tx][i];
}

// ---------------- kernel 1: gating + routing + bf16 cast of x -------------
__global__ __launch_bounds__(256)
void k_gate(const float* __restrict__ x, const float* __restrict__ wg,
            __bf16* __restrict__ xbf, int* __restrict__ counts,
            int* __restrict__ lists, float* __restrict__ slotw,
            float* __restrict__ aux) {
  const int t   = blockIdx.x;
  const int tid = threadIdx.x;          // 256 threads
  __shared__ float red[EEXP][256];

  float p[EEXP];
#pragma unroll
  for (int e = 0; e < EEXP; ++e) p[e] = 0.0f;

  const float* xr = x + (size_t)t * DDIM;
  __bf16*      xb = xbf + (size_t)t * DDIM;
  for (int d = tid; d < DDIM; d += 256) {
    float v = xr[d];
    xb[d] = (__bf16)v;
    const float* wr = wg + (size_t)d * EEXP;
#pragma unroll
    for (int e = 0; e < EEXP; ++e) p[e] += v * wr[e];
  }
#pragma unroll
  for (int e = 0; e < EEXP; ++e) red[e][tid] = p[e];
  __syncthreads();

  if (tid < EEXP) {
    float s = 0.0f;
    for (int i = 0; i < 256; ++i) s += red[tid][i];
    red[tid][0] = s;
  }
  __syncthreads();

  if (tid == 0) {
    float l[EEXP];
    float m = -1e30f, mean = 0.0f;
#pragma unroll
    for (int e = 0; e < EEXP; ++e) {
      l[e] = red[e][0];
      m = fmaxf(m, l[e]);
      mean += l[e];
    }
    mean *= (1.0f / EEXP);
    float var = 0.0f;
#pragma unroll
    for (int e = 0; e < EEXP; ++e) { float d = l[e] - mean; var += d * d; }
    var *= (1.0f / (EEXP - 1));                 // unbiased (ddof=1)
    atomicAdd(aux, var * (1.0f / T_TOK));       // mean over tokens

    float pe[EEXP];
#pragma unroll
    for (int e = 0; e < EEXP; ++e) pe[e] = __expf(l[e] - m);

    int i1 = 0;
    for (int e = 1; e < EEXP; ++e) if (pe[e] > pe[i1]) i1 = e;
    int i2 = (i1 == 0) ? 1 : 0;
    for (int e = 0; e < EEXP; ++e) if (e != i1 && pe[e] > pe[i2]) i2 = e;

    const float inv = 1.0f / (pe[i1] + pe[i2]);
    int s1 = atomicAdd(&counts[i1], 1);
    lists[i1 * T_TOK + s1] = t;  slotw[i1 * T_TOK + s1] = pe[i1] * inv;
    int s2 = atomicAdd(&counts[i2], 1);
    lists[i2 * T_TOK + s2] = t;  slotw[i2 * T_TOK + s2] = pe[i2] * inv;
  }
}

// ---------------- kernel 2: exclusive prefix of counts ----------------
__global__ void k_offsets(const int* __restrict__ counts, int* __restrict__ offs) {
  if (threadIdx.x == 0) {
    int a = 0;
#pragma unroll
    for (int e = 0; e < EEXP; ++e) { offs[e] = a; a += counts[e]; }
  }
}

// ---------------- kernel 3: h = silu(x@W1) * (x@W3), gathered ----------------
__global__ __launch_bounds__(256)
void k_ffn_up(const __bf16* __restrict__ xbf,
              const __bf16* __restrict__ w1t, const __bf16* __restrict__ w3t,
              __bf16* __restrict__ hbuf,
              const int* __restrict__ counts, const int* __restrict__ offs,
              const int* __restrict__ lists) {
  const int e  = blockIdx.x / TILES_M;
  const int mt = blockIdx.x % TILES_M;
  const int cnt = counts[e];
  if (mt * BM >= cnt) return;
  int rows = cnt - mt * BM; if (rows > BM) rows = BM;
  const int n0 = blockIdx.y * BN;

  __shared__ __align__(64) __bf16 As [BM][BK];
  __shared__ __align__(64) __bf16 Bs1[BN][BK];
  __shared__ __align__(64) __bf16 Bs3[BN][BK];
  __shared__ int toks[BM];

  const int tid = threadIdx.x;
  if (tid < BM) {
    int lr = tid < rows ? tid : rows - 1;       // clamp: duplicated compute, never stored
    toks[tid] = lists[e * T_TOK + mt * BM + lr];
  }
  __syncthreads();

  const int wave = tid >> 5, lane = tid & 31;
  const int mBase = (wave >> 1) * 32;           // 4 waves along M
  const int nBase = (wave & 1)  * 32;           // 2 waves along N
  const int l16 = lane & 15, hi = lane >> 4;

  v8f acc1[2][2] = {{{0}}, {{0}}};
  v8f acc3[2][2] = {{{0}}, {{0}}};

  const __bf16* w1e = w1t + (size_t)e * DDIM * FDIM;   // [F, D] n-major bf16
  const __bf16* w3e = w3t + (size_t)e * DDIM * FDIM;

  const int arow  = tid >> 1;                   // A staging: 128 rows x 32 k, 32B/thread
  const int ahalf = (tid & 1) * 16;

  for (int k0 = 0; k0 < DDIM; k0 += BK) {
    // A: gathered bf16 token rows (manual — irregular rows)
    *(v16bf*)&As[arow][ahalf] =
        *(const v16bf*)(xbf + (size_t)toks[arow] * DDIM + k0 + ahalf);
    // B tiles via Tensor Data Mover: [BN rows x BK elems], row stride D
    if (wave == 0)
      tdm_load_2d(lds_byte_addr(&Bs1[0][0]), w1e + (size_t)n0 * DDIM + k0,
                  DDIM, FDIM, BK, BN, DDIM);
    else if (wave == 1)
      tdm_load_2d(lds_byte_addr(&Bs3[0][0]), w3e + (size_t)n0 * DDIM + k0,
                  DDIM, FDIM, BK, BN, DDIM);
    __builtin_amdgcn_s_wait_tensorcnt(0);
    __syncthreads();

    v16bf a[2], b1[2], b3[2];
#pragma unroll
    for (int mi = 0; mi < 2; ++mi) {            // A frag: 16 contiguous K per lane
      const int r = mBase + mi * 16 + l16, kb = hi * 16;
      *(v8bf*)&a[mi]       = *(const v8bf*)&As[r][kb];
      *((v8bf*)&a[mi] + 1) = *(const v8bf*)&As[r][kb + 8];
    }
#pragma unroll
    for (int ni = 0; ni < 2; ++ni) {            // B frag: 16 contiguous K per lane
      const int n = nBase + ni * 16 + l16, kb = hi * 16;
      b1[ni] = *(const v16bf*)&Bs1[n][kb];
      b3[ni] = *(const v16bf*)&Bs3[n][kb];
    }
#pragma unroll
    for (int mi = 0; mi < 2; ++mi)
#pragma unroll
      for (int ni = 0; ni < 2; ++ni) {
        acc1[mi][ni] = __builtin_amdgcn_wmma_f32_16x16x32_bf16(
            false, a[mi], false, b1[ni], (short)0, acc1[mi][ni], false, false);
        acc3[mi][ni] = __builtin_amdgcn_wmma_f32_16x16x32_bf16(
            false, a[mi], false, b3[ni], (short)0, acc3[mi][ni], false, false);
      }
    __syncthreads();
  }

  // epilogue: silu(u) * v -> bf16 h rows (compact layout)
  const size_t hRowBase = (size_t)(offs[e] + mt * BM);
#pragma unroll
  for (int mi = 0; mi < 2; ++mi)
#pragma unroll
    for (int ni = 0; ni < 2; ++ni)
#pragma unroll
      for (int r = 0; r < 8; ++r) {
        const int lm = mBase + mi * 16 + r + hi * 8;   // C layout: M = r + 8*(lane>=16)
        if (lm < rows) {
          const float u = acc1[mi][ni][r];
          const float g = u / (1.0f + __expf(-u));
          const float hv = g * acc3[mi][ni][r];
          const int col = n0 + nBase + ni * 16 + l16;
          hbuf[(hRowBase + lm) * (size_t)FDIM + col] = (__bf16)hv;
        }
      }
}

// ---------------- kernel 4: out += weight * (h @ W2[e]), scattered ----------------
__global__ __launch_bounds__(256)
void k_ffn_down(const __bf16* __restrict__ hbuf, const __bf16* __restrict__ w2t,
                float* __restrict__ out,
                const int* __restrict__ counts, const int* __restrict__ offs,
                const int* __restrict__ lists, const float* __restrict__ slotw) {
  const int e  = blockIdx.x / TILES_M;
  const int mt = blockIdx.x % TILES_M;
  const int cnt = counts[e];
  if (mt * BM >= cnt) return;
  int rows = cnt - mt * BM; if (rows > BM) rows = BM;
  const int n0 = blockIdx.y * BN;               // over D

  __shared__ __align__(64) __bf16 As[BM][BK];
  __shared__ __align__(64) __bf16 Bs[BN][BK];

  const int tid = threadIdx.x;
  const int wave = tid >> 5, lane = tid & 31;
  const int mBase = (wave >> 1) * 32;
  const int nBase = (wave & 1)  * 32;
  const int l16 = lane & 15, hi = lane >> 4;

  v8f acc[2][2] = {{{0}}, {{0}}};

  const __bf16* hrow = hbuf + (size_t)(offs[e] + mt * BM) * FDIM;  // [rows, F]
  const __bf16* w2e  = w2t + (size_t)e * DDIM * FDIM;              // [D, F] n-major

  for (int k0 = 0; k0 < FDIM; k0 += BK) {
    // Both tiles via TDM. A: tensor_dim1 = rows -> zero-fill past expert rows.
    if (wave == 0)
      tdm_load_2d(lds_byte_addr(&As[0][0]), hrow + k0,
                  FDIM, (unsigned)rows, BK, BM, FDIM);
    else if (wave == 1)
      tdm_load_2d(lds_byte_addr(&Bs[0][0]), w2e + (size_t)n0 * FDIM + k0,
                  FDIM, DDIM, BK, BN, FDIM);
    __builtin_amdgcn_s_wait_tensorcnt(0);
    __syncthreads();

    v16bf a[2], b[2];
#pragma unroll
    for (int mi = 0; mi < 2; ++mi) {
      const int r = mBase + mi * 16 + l16, kb = hi * 16;
      *(v8bf*)&a[mi]       = *(const v8bf*)&As[r][kb];
      *((v8bf*)&a[mi] + 1) = *(const v8bf*)&As[r][kb + 8];
    }
#pragma unroll
    for (int ni = 0; ni < 2; ++ni) {
      const int n = nBase + ni * 16 + l16, kb = hi * 16;
      b[ni] = *(const v16bf*)&Bs[n][kb];
    }
#pragma unroll
    for (int mi = 0; mi < 2; ++mi)
#pragma unroll
      for (int ni = 0; ni < 2; ++ni)
        acc[mi][ni] = __builtin_amdgcn_wmma_f32_16x16x32_bf16(
            false, a[mi], false, b[ni], (short)0, acc[mi][ni], false, false);
    __syncthreads();
  }

  // epilogue: scale by routing weight, atomically merge the K=2 expert outputs
#pragma unroll
  for (int mi = 0; mi < 2; ++mi)
#pragma unroll
    for (int ni = 0; ni < 2; ++ni)
#pragma unroll
      for (int r = 0; r < 8; ++r) {
        const int lm = mBase + mi * 16 + r + hi * 8;
        if (lm < rows) {
          const int idx = e * T_TOK + mt * BM + lm;
          const int tok = lists[idx];
          const float w = slotw[idx];
          const int col = n0 + nBase + ni * 16 + l16;
          atomicAdd(&out[(size_t)tok * DDIM + col], w * acc[mi][ni][r]);
        }
      }
}

// ---------------- host launcher ----------------
extern "C" void kernel_launch(void* const* d_in, const int* in_sizes, int n_in,
                              void* d_out, int out_size, void* d_ws, size_t ws_size,
                              hipStream_t stream) {
  const float* x  = (const float*)d_in[0];   // [B,S,D]
  const float* Wg = (const float*)d_in[1];   // [D,E]
  const float* W1 = (const float*)d_in[2];   // [E,D,F]
  const float* W2 = (const float*)d_in[3];   // [E,F,D]
  const float* W3 = (const float*)d_in[4];   // [E,D,F]
  float* out = (float*)d_out;                // [B,S,D] ++ [aux_loss]

  char* ws = (char*)d_ws;
  __bf16* xbf   = (__bf16*)(ws + WS_XBF);
  __bf16* hbuf  = (__bf16*)(ws + WS_HBUF);
  __bf16* w1t   = (__bf16*)(ws + WS_W1T);
  __bf16* w3t   = (__bf16*)(ws + WS_W3T);
  __bf16* w2t   = (__bf16*)(ws + WS_W2T);
  int*    cnts  = (int*)   (ws + WS_CNT);
  int*    offs  = (int*)   (ws + WS_OFF);
  int*    lists = (int*)   (ws + WS_LIST);
  float*  slotw = (float*) (ws + WS_SLOTW);
  float*  aux   = out + (size_t)T_TOK * DDIM;

  k_zero<<<1024, 256, 0, stream>>>(out, out_size, cnts);

  // one-shot weight cast+transpose: fp32 [R,C] -> bf16 [C,R]
  dim3 tb(32, 8);
  k_transpose_cast<<<dim3(FDIM / 32, DDIM / 32, EEXP), tb, 0, stream>>>(
      W1, w1t, DDIM, FDIM);   // [D,F] -> [F,D]
  k_transpose_cast<<<dim3(FDIM / 32, DDIM / 32, EEXP), tb, 0, stream>>>(
      W3, w3t, DDIM, FDIM);   // [D,F] -> [F,D]
  k_transpose_cast<<<dim3(DDIM / 32, FDIM / 32, EEXP), tb, 0, stream>>>(
      W2, w2t, FDIM, DDIM);   // [F,D] -> [D,F]

  k_gate<<<T_TOK, 256, 0, stream>>>(x, Wg, xbf, cnts, lists, slotw, aux);
  k_offsets<<<1, 32, 0, stream>>>(cnts, offs);
  k_ffn_up<<<dim3(EEXP * TILES_M, FDIM / BN), 256, 0, stream>>>(
      xbf, w1t, w3t, hbuf, cnts, offs, lists);
  k_ffn_down<<<dim3(EEXP * TILES_M, DDIM / BN), 256, 0, stream>>>(
      hbuf, w2t, out, cnts, offs, lists, slotw);
}
